// Arch8BLayer_44435731645127
// MI455X (gfx1250) — compile-verified, matrix operand order checked by reading, not run on hardware
//
#include <hip/hip_runtime.h>
#include <hip/hip_bf16.h>

// ---------------------------------------------------------------------------
// Fixed problem sizes (from reference setup_inputs)
// ---------------------------------------------------------------------------
#define NROWS   131072      // n = S*k
#define HD      256         // hidden dim
#define SSUBS   4096        // S
#define KSUB    32          // k
#define NTOT    16384       // N_total
#define NEDGE   262144      // E (global edges)
#define NEI     262144      // E_intra
#define NHEADS  8
#define HEADD   32          // HD / NHEADS

typedef __attribute__((ext_vector_type(16))) __bf16 v16bf;
typedef __attribute__((ext_vector_type(8)))  __bf16 v8bf;
typedef __attribute__((ext_vector_type(8)))  float  v8f;

// ---------------------------------------------------------------------------
// BFS distances (Bellman-Ford with atomicMin; 31 in-place rounds == 31
// synchronous rounds of the reference: paths in a 32-node subgraph <= 31 hops)
// ---------------------------------------------------------------------------
__global__ void dist_init_k(int* dist) {
    int i = blockIdx.x * blockDim.x + threadIdx.x;
    if (i < NROWS) dist[i] = ((i & (KSUB - 1)) == 0) ? 0 : KSUB;
}

__global__ void dist_relax_k(const int* __restrict__ ei, int* __restrict__ dist) {
    int e = blockIdx.x * blockDim.x + threadIdx.x;
    if (e >= 2 * NEI) return;
    int src = ei[e];                                   // row0 fwd / row1 bwd
    int dst = (e < NEI) ? ei[e + NEI] : ei[e - NEI];
    int d = dist[src] + 1;
    if (d > KSUB) d = KSUB;
    atomicMin(&dist[dst], d);
}

// hin = h_flat + dist_emb[dist]   (dist in [0, 32], dist_emb has 33 rows)
__global__ void add_dist_emb_k(const float* __restrict__ h, const int* __restrict__ dist,
                               const float* __restrict__ emb, float* __restrict__ hin) {
    size_t i = blockIdx.x; int c = threadIdx.x;
    int d = dist[i];
    hin[i * HD + c] = h[i * HD + c] + emb[(size_t)d * HD + c];
}

// ---------------------------------------------------------------------------
// Row LayerNorm over HD=256 (one block per row)
// ---------------------------------------------------------------------------
__global__ void layernorm_k(const float* __restrict__ X, const float* __restrict__ g,
                            const float* __restrict__ b, float* __restrict__ Y) {
    __shared__ float red[HD];
    size_t row = blockIdx.x; int c = threadIdx.x;
    float x = X[row * HD + c];
    red[c] = x; __syncthreads();
    for (int s = HD / 2; s > 0; s >>= 1) { if (c < s) red[c] += red[c + s]; __syncthreads(); }
    float m = red[0] / (float)HD;
    __syncthreads();
    float d = x - m;
    red[c] = d * d; __syncthreads();
    for (int s = HD / 2; s > 0; s >>= 1) { if (c < s) red[c] += red[c + s]; __syncthreads(); }
    float var = red[0] / (float)HD;
    Y[row * HD + c] = d * rsqrtf(var + 1e-5f) * g[c] + b[c];
}

// ---------------------------------------------------------------------------
// Block-tiled WMMA bf16 GEMM: C[rows x cols] = A[rows x K] @ W[K x cols] + bias
// BM=128 x BN=64 x BK=32 per block (8 waves). A/B staged through LDS with
// f32->bf16 conversion; B tile shared by all waves. Software-pipelined:
// next K-step's global loads are issued while WMMA consumes the current LDS
// tile, hiding global latency behind matrix ops.
//
// LDS layout: row-major bf16, padded row stride 40 bf16 (80 B = 20 dwords) so
// the two 8-element K-runs of each lane's WMMA fragment are single
// 16B-aligned ds_load_b128 ops with conflict-minimal banking.
//
// ISA 16-bit A-matrix 16x32 layout: lane half h=lane>>4, r=lane&15; fragment
// element e<8 -> K = 8h+e, e>=8 -> K = 16+8h+(e-8)  (two contiguous runs).
// C/D layout: element v -> M = v + 8*h, N = r.
// ---------------------------------------------------------------------------
#define BM 128
#define BN 64
#define BK 32
#define LPAD 40   // bf16 elems per padded LDS row (80 bytes)

__device__ __forceinline__ v8bf pack8(float4 a, float4 b) {
    v8bf o;
    o[0] = (__bf16)a.x; o[1] = (__bf16)a.y; o[2] = (__bf16)a.z; o[3] = (__bf16)a.w;
    o[4] = (__bf16)b.x; o[5] = (__bf16)b.y; o[6] = (__bf16)b.z; o[7] = (__bf16)b.w;
    return o;
}

template <bool RELU, bool HAS_RES>
__global__ void gemm_wmma_k(const float* __restrict__ A, const float* __restrict__ W,
                            const float* __restrict__ bias, const float* __restrict__ residual,
                            float* __restrict__ C, int rows, int K, int cols) {
    __shared__ __align__(16) __bf16 At[BM * LPAD];   // [row][k]
    __shared__ __align__(16) __bf16 Bt[BN * LPAD];   // [col][k]

    const int tid  = threadIdx.x;
    const int wave = tid >> 5;
    const int lane = tid & 31;
    const int h = lane >> 4;
    const int r = lane & 15;

    const int colBlocks = cols >> 6;
    const int rowBase = (blockIdx.x / colBlocks) * BM;
    const int colBase = (blockIdx.x % colBlocks) * BN;

    // staging assignments
    const int arow  = tid >> 1;        // 0..127
    const int ahalf = tid & 1;         // 16 floats each
    const int bcol  = tid & 63;        // 0..63
    const int bq    = tid >> 6;        // 0..3, 8 K values each

    const float* aptr = A + (size_t)(rowBase + arow) * K + ahalf * 16;
    const float* wptr = W + (size_t)(bq * 8) * cols + colBase + bcol;

    v8f zero = {};
    v8f acc[4];
    #pragma unroll
    for (int i = 0; i < 4; ++i) acc[i] = zero;

    // ---- prologue: load first K-step into registers ----
    float4 a0, a1, a2, a3;
    float  wv[8];
    {
        const float4* ap = (const float4*)aptr;
        a0 = ap[0]; a1 = ap[1]; a2 = ap[2]; a3 = ap[3];
        #pragma unroll
        for (int i = 0; i < 8; ++i) wv[i] = wptr[(size_t)i * cols];
    }

    for (int kb = 0; kb < K; kb += BK) {
        // ---- commit staged registers to LDS (f32 -> bf16) ----
        *(v8bf*)(&At[arow * LPAD + ahalf * 16])     = pack8(a0, a1);
        *(v8bf*)(&At[arow * LPAD + ahalf * 16 + 8]) = pack8(a2, a3);
        *(v8bf*)(&Bt[bcol * LPAD + bq * 8]) =
            pack8(make_float4(wv[0], wv[1], wv[2], wv[3]),
                  make_float4(wv[4], wv[5], wv[6], wv[7]));
        __syncthreads();

        // ---- issue next K-step's global loads (overlap with WMMA below) ----
        if (kb + BK < K) {
            const float4* ap = (const float4*)(aptr + kb + BK);
            a0 = ap[0]; a1 = ap[1]; a2 = ap[2]; a3 = ap[3];
            const float* wp = wptr + (size_t)(kb + BK) * cols;
            #pragma unroll
            for (int i = 0; i < 8; ++i) wv[i] = wp[(size_t)i * cols];
        }

        // ---- compute: each wave 16 rows x 64 cols ----
        const int trow = wave * 16 + r;
        v8bf alo = *(const v8bf*)(&At[trow * LPAD + 8 * h]);
        v8bf ahi = *(const v8bf*)(&At[trow * LPAD + 16 + 8 * h]);
        v16bf afrag = __builtin_shufflevector(alo, ahi,
            0, 1, 2, 3, 4, 5, 6, 7, 8, 9, 10, 11, 12, 13, 14, 15);
        #pragma unroll
        for (int cg = 0; cg < 4; ++cg) {
            const int col = cg * 16 + r;
            v8bf blo = *(const v8bf*)(&Bt[col * LPAD + 8 * h]);
            v8bf bhi = *(const v8bf*)(&Bt[col * LPAD + 16 + 8 * h]);
            v16bf bfrag = __builtin_shufflevector(blo, bhi,
                0, 1, 2, 3, 4, 5, 6, 7, 8, 9, 10, 11, 12, 13, 14, 15);
            acc[cg] = __builtin_amdgcn_wmma_f32_16x16x32_bf16(
                false, afrag, false, bfrag, (short)0, acc[cg], false, false);
        }
        __syncthreads();
    }

    // ---- epilogue (branch-free; RELU/HAS_RES are compile-time) ----
    #pragma unroll
    for (int cg = 0; cg < 4; ++cg) {
        const int col = colBase + cg * 16 + r;
        const float bsc = bias[col];
        const size_t base = (size_t)(rowBase + wave * 16 + 8 * h) * cols + col;
        #pragma unroll
        for (int v = 0; v < 8; ++v) {
            size_t idx = base + (size_t)v * cols;
            float val = acc[cg][v] + bsc;
            if (HAS_RES) val += residual[idx];
            if (RELU) val = fmaxf(val, 0.f);
            C[idx] = val;
        }
    }
}

// ---------------------------------------------------------------------------
// Per-subgraph masked multi-head attention (k=32, 8 heads, hd=32).
// One block per subgraph; thread = (query, head). K/V staged in 64KB LDS.
// ---------------------------------------------------------------------------
__global__ void attention_k(const float* __restrict__ q, const float* __restrict__ kk,
                            const float* __restrict__ v, const unsigned char* __restrict__ valid,
                            float* __restrict__ o) {
    __shared__ float Ks[KSUB][HD];
    __shared__ float Vs[KSUB][HD];
    __shared__ float pad[KSUB];
    int s = blockIdx.x, tid = threadIdx.x;
    size_t base = (size_t)s * KSUB * HD;
    for (int t = tid; t < KSUB * HD; t += 256) {
        Ks[t >> 8][t & 255] = kk[base + t];
        Vs[t >> 8][t & 255] = v[base + t];
    }
    if (tid < KSUB) pad[tid] = valid[(size_t)s * KSUB + tid] ? 0.f : 1.f;
    __syncthreads();

    int qi = tid >> 3, head = tid & 7, ho = head * HEADD;
    float qv[HEADD];
    const float* qrow = q + base + (size_t)qi * HD + ho;
    #pragma unroll
    for (int d = 0; d < HEADD; ++d) qv[d] = qrow[d];

    float sc[KSUB]; float mx = -1e30f;
    #pragma unroll
    for (int j = 0; j < KSUB; ++j) {
        float dot = 0.f;
        #pragma unroll
        for (int d = 0; d < HEADD; ++d) dot += qv[d] * Ks[j][ho + d];
        dot *= 0.17677669529663687f;               // 1/sqrt(32)
        if (pad[j] > 0.5f) dot = -1e9f;
        sc[j] = dot; mx = fmaxf(mx, dot);
    }
    float sum = 0.f;
    #pragma unroll
    for (int j = 0; j < KSUB; ++j) { sc[j] = __expf(sc[j] - mx); sum += sc[j]; }
    float inv = 1.f / sum;
    float acc[HEADD];
    #pragma unroll
    for (int d = 0; d < HEADD; ++d) acc[d] = 0.f;
    #pragma unroll
    for (int j = 0; j < KSUB; ++j) {
        float w = sc[j] * inv;
        #pragma unroll
        for (int d = 0; d < HEADD; ++d) acc[d] += w * Vs[j][ho + d];
    }
    float* orow = o + base + (size_t)qi * HD + ho;
    #pragma unroll
    for (int d = 0; d < HEADD; ++d) orow[d] = acc[d];
}

// ---------------------------------------------------------------------------
// BatchNorm: per-channel sum/sumsq (atomics), finalize to scale/shift, apply.
// ---------------------------------------------------------------------------
__global__ void bn_stats_k(const float* __restrict__ X, const unsigned char* __restrict__ mask,
                           int rows_per_block, int rows, float* __restrict__ stats) {
    int c = threadIdx.x;
    int r0 = blockIdx.x * rows_per_block;
    int r1 = min(r0 + rows_per_block, rows);
    float s = 0.f, sq = 0.f;
    for (int r = r0; r < r1; ++r) {
        float x = X[(size_t)r * HD + c];
        if (mask && !mask[r]) x = 0.f;
        s += x; sq += x * x;
    }
    atomicAdd(&stats[c], s);
    atomicAdd(&stats[HD + c], sq);
}

__global__ void bn_finalize_k(float* __restrict__ stats, const float* __restrict__ g,
                              const float* __restrict__ b, float rows) {
    int c = threadIdx.x;
    float m = stats[c] / rows;
    float var = stats[HD + c] / rows - m * m;
    float a = g[c] * rsqrtf(var + 1e-5f);
    stats[c] = a;
    stats[HD + c] = b[c] - m * a;
}

__global__ void bn_apply_k(float* __restrict__ X, const float* __restrict__ stats) {
    size_t i = blockIdx.x; int c = threadIdx.x;
    X[i * HD + c] = X[i * HD + c] * stats[c] + stats[HD + c];
}

// ---------------------------------------------------------------------------
// Global scatter-mean over node_ids, GINE edge messages, g = (1+eps)x + aggr
// ---------------------------------------------------------------------------
__global__ void scatter_accum_k(const float* __restrict__ h, const unsigned char* __restrict__ valid,
                                const int* __restrict__ ids, float* __restrict__ gsum,
                                float* __restrict__ gcnt) {
    size_t i = blockIdx.x; int c = threadIdx.x;
    if (!valid[i]) return;
    int id = ids[i]; if (id < 0) id = 0;
    atomicAdd(&gsum[(size_t)id * HD + c], h[i * HD + c]);
    if (c == 0) atomicAdd(&gcnt[id], 1.f);
}

__global__ void xsum_div_k(float* __restrict__ gsum, const float* __restrict__ gcnt) {
    size_t i = blockIdx.x; int c = threadIdx.x;
    gsum[i * HD + c] /= fmaxf(gcnt[i], 1.f);
}

__global__ void edge_msg_k(const float* __restrict__ ea, const int* __restrict__ eidx,
                           const float* __restrict__ W, const float* __restrict__ b,
                           const float* __restrict__ xsum, float* __restrict__ aggr) {
    __shared__ float eas[32];
    int e = blockIdx.x, c = threadIdx.x;
    if (c < 32) eas[c] = ea[(size_t)e * 32 + c];
    __syncthreads();
    int src = eidx[e];
    int dst = eidx[NEDGE + e];
    float acc = b[c];
    #pragma unroll
    for (int j = 0; j < 32; ++j) acc += eas[j] * W[j * HD + c];
    float msg = fmaxf(xsum[(size_t)src * HD + c] + acc, 0.f);
    atomicAdd(&aggr[(size_t)dst * HD + c], msg);
}

__global__ void make_g_k(const float* __restrict__ xsum, float* __restrict__ aggr,
                         const float* __restrict__ eps) {
    size_t i = blockIdx.x; int c = threadIdx.x;
    aggr[i * HD + c] = (1.f + eps[0]) * xsum[i * HD + c] + aggr[i * HD + c];
}

// subgraph masked mean (sub_batch is contiguous groups of k=32 rows)
__global__ void submean_k(const float* __restrict__ h, const unsigned char* __restrict__ valid,
                          float* __restrict__ hsub) {
    size_t s = blockIdx.x; int c = threadIdx.x;
    float sum = 0.f, cnt = 0.f;
    #pragma unroll
    for (int r = 0; r < KSUB; ++r) {
        if (valid[s * KSUB + r]) { sum += h[(s * KSUB + r) * HD + c]; cnt += 1.f; }
    }
    hsub[s * HD + c] = sum / fmaxf(cnt, 1.f);
}

// out = relu(skip + h1 + h2[ids] + hsub[i/k]) * valid
__global__ void final_k(const float* __restrict__ skip, const float* __restrict__ h1,
                        const float* __restrict__ h2, const float* __restrict__ hsub,
                        const int* __restrict__ ids, const unsigned char* __restrict__ valid,
                        float* __restrict__ out) {
    size_t i = blockIdx.x; int c = threadIdx.x;
    int id = ids[i]; if (id < 0) id = 0;
    float val = skip[i * HD + c] + h1[i * HD + c]
              + h2[(size_t)id * HD + c] + hsub[(i >> 5) * HD + c];
    out[i * HD + c] = valid[i] ? fmaxf(val, 0.f) : 0.f;
}

// ---------------------------------------------------------------------------
// Launch sequence
// ---------------------------------------------------------------------------
static void launch_gemm(const float* A, const float* W, const float* bias, const float* res,
                        float* C, int rows, int K, int cols, int relu, hipStream_t stream) {
    int blocks = (rows / BM) * (cols / BN);
    if (res)
        gemm_wmma_k<false, true><<<blocks, 256, 0, stream>>>(A, W, bias, res, C, rows, K, cols);
    else if (relu)
        gemm_wmma_k<true, false><<<blocks, 256, 0, stream>>>(A, W, bias, nullptr, C, rows, K, cols);
    else
        gemm_wmma_k<false, false><<<blocks, 256, 0, stream>>>(A, W, bias, nullptr, C, rows, K, cols);
}

extern "C" void kernel_launch(void* const* d_in, const int* in_sizes, int n_in,
                              void* d_out, int out_size, void* d_ws, size_t ws_size,
                              hipStream_t stream) {
    // ---- inputs (setup_inputs dict order; params dict flattened in order) ----
    const float* h_flat  = (const float*)d_in[0];
    const int*   intra_ei= (const int*)  d_in[1];
    // d_in[2] = ea_flat (unused by reference forward)
    const unsigned char* valid = (const unsigned char*)d_in[3];  // jnp bool -> 1 byte
    const int*   node_ids= (const int*)  d_in[4];
    // d_in[5] = N_total (scalar, hardcoded)
    const int*   edge_idx= (const int*)  d_in[6];
    const float* edge_attr=(const float*)d_in[7];
    // d_in[8] = sub_batch (contiguous groups, hardcoded), 9=S, 10=k
    const float* dist_emb= (const float*)d_in[11];
    const float* Wq = (const float*)d_in[12];
    const float* Wk = (const float*)d_in[13];
    const float* Wv = (const float*)d_in[14];
    const float* Wo = (const float*)d_in[15];
    const float* ffW1=(const float*)d_in[16];
    const float* ffW2=(const float*)d_in[17];
    const float* edgeW=(const float*)d_in[18];
    const float* gW1 =(const float*)d_in[19];
    const float* gW2 =(const float*)d_in[20];
    const float* skW =(const float*)d_in[21];
    const float* sW1 =(const float*)d_in[22];
    const float* sW2 =(const float*)d_in[23];
    const float* bq=(const float*)d_in[24], *bk=(const float*)d_in[25];
    const float* bv=(const float*)d_in[26], *bo=(const float*)d_in[27];
    const float* ffb2=(const float*)d_in[28], *edgeb=(const float*)d_in[29];
    const float* gb1=(const float*)d_in[30], *gb2=(const float*)d_in[31];
    const float* skb=(const float*)d_in[32], *sb1=(const float*)d_in[33];
    const float* sb2=(const float*)d_in[34];
    const float* ln1b=(const float*)d_in[35], *ln2b=(const float*)d_in[36];
    const float* subbnb=(const float*)d_in[37], *gbnb=(const float*)d_in[38];
    const float* sbnb=(const float*)d_in[39];
    const float* ffb1=(const float*)d_in[40];
    const float* ln1g=(const float*)d_in[41], *ln2g=(const float*)d_in[42];
    const float* subbng=(const float*)d_in[43], *gbng=(const float*)d_in[44];
    const float* sbng=(const float*)d_in[45];
    const float* gine_eps=(const float*)d_in[46];
    float* out = (float*)d_out;

    // ---- workspace layout (needs 6 * n*H floats = ~805 MB) ----
    const size_t NH = (size_t)NROWS * HD;
    float* ws  = (float*)d_ws;
    float* hin = ws;                 // B0: hin, later f=LN(x)
    float* B1  = ws + NH;            // a, later o, later h_skip
    float* B2  = ws + 2 * NH;        // q, later x2 -> h1 (in place)
    float* B3  = ws + 3 * NH;        // kbuf; later ffmid (2*NH, spans B3+B4); later G region
    float* B4  = ws + 4 * NH;        // vbuf
    float* B5  = ws + 5 * NH;        // x
    int*   dist = (int*)B2;          // B2 unused until q is written
    const size_t NTH = (size_t)NTOT * HD;
    const size_t SH  = (size_t)SSUBS * HD;
    float* Gsum  = B3;               // scatter sums -> x_sum (in place)
    float* Gcnt  = B3 + NTH;
    float* Gaggr = Gcnt + NTOT;      // aggr -> g (in place)
    float* Gmid  = Gaggr + NTH;
    float* Gh2   = Gmid + NTH;
    float* Ssub  = Gh2 + NTH;
    float* Smid  = Ssub + SH;
    float* Ssub2 = Smid + SH;
    float* stA   = Ssub2 + SH;       // 512 floats each
    float* stB   = stA + 2 * HD;
    float* stC   = stB + 2 * HD;

    // 1) BFS distances
    dist_init_k<<<(NROWS + 255) / 256, 256, 0, stream>>>(dist);
    for (int it = 0; it < KSUB - 1; ++it)
        dist_relax_k<<<(2 * NEI + 255) / 256, 256, 0, stream>>>(intra_ei, dist);
    // 2) hin = h + dist_emb[dist]
    add_dist_emb_k<<<NROWS, HD, 0, stream>>>(h_flat, dist, dist_emb, hin);
    // 3) a = LN(hin)
    layernorm_k<<<NROWS, HD, 0, stream>>>(hin, ln1g, ln1b, B1);
    // 4) q,k,v GEMMs (WMMA bf16)
    launch_gemm(B1, Wq, bq, nullptr, B2, NROWS, HD, HD, 0, stream);
    launch_gemm(B1, Wk, bk, nullptr, B3, NROWS, HD, HD, 0, stream);
    launch_gemm(B1, Wv, bv, nullptr, B4, NROWS, HD, HD, 0, stream);
    // 5) attention -> o (B1)
    attention_k<<<SSUBS, 256, 0, stream>>>(B2, B3, B4, valid, B1);
    // 6) x = hin + o@Wo + bo
    launch_gemm(B1, Wo, bo, hin, B5, NROWS, HD, HD, 0, stream);
    // 7) f = LN(x) -> B0
    layernorm_k<<<NROWS, HD, 0, stream>>>(B5, ln2g, ln2b, hin);
    // 8) ffmid = relu(f@W1+b1) -> B3 (2*NH); 9) x2 = x + ffmid@W2+b2 -> B2
    launch_gemm(hin, ffW1, ffb1, nullptr, B3, NROWS, HD, 2 * HD, 1, stream);
    launch_gemm(B3, ffW2, ffb2, B5, B2, NROWS, 2 * HD, HD, 0, stream);
    // 10) zero scratch that needs it (ffmid now dead)
    hipMemsetAsync(Gsum,  0, NTH * sizeof(float), stream);
    hipMemsetAsync(Gcnt,  0, NTOT * sizeof(float), stream);
    hipMemsetAsync(Gaggr, 0, NTH * sizeof(float), stream);
    hipMemsetAsync(stA,   0, 6 * HD * sizeof(float), stream);
    // 11) sub_bn: stats over x2*valid across all n rows; apply in place -> h1 = B2
    bn_stats_k<<<NROWS / 256, HD, 0, stream>>>(B2, valid, 256, NROWS, stA);
    bn_finalize_k<<<1, HD, 0, stream>>>(stA, subbng, subbnb, (float)NROWS);
    bn_apply_k<<<NROWS, HD, 0, stream>>>(B2, stA);
    // 12) h_skip = h@skW + skb -> B1 (o dead)
    launch_gemm(h_flat, skW, skb, nullptr, B1, NROWS, HD, HD, 0, stream);
    // 13) global scatter-mean -> x_sum
    scatter_accum_k<<<NROWS, HD, 0, stream>>>(h_flat, valid, node_ids, Gsum, Gcnt);
    xsum_div_k<<<NTOT, HD, 0, stream>>>(Gsum, Gcnt);
    // 14) GINE messages: aggr += relu(x_sum[src] + ea@edgeW + edgeb)
    edge_msg_k<<<NEDGE, HD, 0, stream>>>(edge_attr, edge_idx, edgeW, edgeb, Gsum, Gaggr);
    // 15) g = (1+eps)*x_sum + aggr (in place over aggr); gmlp; gbn
    make_g_k<<<NTOT, HD, 0, stream>>>(Gsum, Gaggr, gine_eps);
    launch_gemm(Gaggr, gW1, gb1, nullptr, Gmid, NTOT, HD, HD, 1, stream);
    launch_gemm(Gmid,  gW2, gb2, nullptr, Gh2,  NTOT, HD, HD, 0, stream);
    bn_stats_k<<<NTOT / 256, HD, 0, stream>>>(Gh2, nullptr, 256, NTOT, stB);
    bn_finalize_k<<<1, HD, 0, stream>>>(stB, gbng, gbnb, (float)NTOT);
    bn_apply_k<<<NTOT, HD, 0, stream>>>(Gh2, stB);
    // 16) subgraph mean readout; smlp; sbn
    submean_k<<<SSUBS, HD, 0, stream>>>(h_flat, valid, Ssub);
    launch_gemm(Ssub, sW1, sb1, nullptr, Smid,  SSUBS, HD, HD, 1, stream);
    launch_gemm(Smid, sW2, sb2, nullptr, Ssub2, SSUBS, HD, HD, 0, stream);
    bn_stats_k<<<SSUBS / 256, HD, 0, stream>>>(Ssub2, nullptr, 256, SSUBS, stC);
    bn_finalize_k<<<1, HD, 0, stream>>>(stC, sbng, sbnb, (float)SSUBS);
    bn_apply_k<<<SSUBS, HD, 0, stream>>>(Ssub2, stC);
    // 17) combine
    final_k<<<NROWS, HD, 0, stream>>>(B1, B2, Gh2, Ssub2, node_ids, valid, out);
    (void)in_sizes; (void)n_in; (void)out_size; (void)ws_size;
}
